// TripletRankingLoss_42365557408212
// MI455X (gfx1250) — 8-bit matrix operand order repaired and checked by reading, compiled, not run on hardware
//
#include <hip/hip_runtime.h>
#include <hip/hip_bf16.h>

typedef __attribute__((ext_vector_type(8))) int v8i;
typedef unsigned long long u64;

#define B_ROWS 8192
#define D_DIM  1024
#define C_CLS  14
#define MARGIN 0.1f
#define EPS    1e-6f

__device__ __forceinline__ unsigned int mix32(unsigned int x) {
    x ^= x >> 16; x *= 0x7feb352dU;
    x ^= x >> 15; x *= 0x846ca68bU;
    x ^= x >> 16;
    return x;
}

__device__ __forceinline__ u64 shfl_xor_u64(u64 v, int mask, int width) {
    unsigned int lo = (unsigned int)v;
    unsigned int hi = (unsigned int)(v >> 32);
    lo = (unsigned int)__shfl_xor((int)lo, mask, width);
    hi = (unsigned int)__shfl_xor((int)hi, mask, width);
    return ((u64)hi << 32) | lo;
}

__global__ __launch_bounds__(128) void trl_phase1_prep(
    const float* __restrict__ pred, const float* __restrict__ gt,
    uint4* __restrict__ gtb, unsigned char* __restrict__ has,
    float* __restrict__ norms)
{
    const int lane = threadIdx.x & 31;
    const int row  = blockIdx.x * 4 + (threadIdx.x >> 5);

    const float4* p = (const float4*)(pred + (size_t)row * D_DIM);
    float s = 0.f;
#pragma unroll
    for (int k = 0; k < 8; ++k) {
        float4 v = p[lane + 32 * k];
        s += v.x * v.x + v.y * v.y + v.z * v.z + v.w * v.w;
    }
#pragma unroll
    for (int off = 16; off; off >>= 1) s += __shfl_xor(s, off, 32);

    if (lane == 0) {
        norms[row] = sqrtf(s);
        unsigned char bytes[16];
        unsigned int any = 0;
#pragma unroll
        for (int c = 0; c < 16; ++c) {
            unsigned char bb = 0;
            if (c < C_CLS) bb = (gt[row * C_CLS + c] > 0.5f) ? 1u : 0u;
            bytes[c] = bb; any |= bb;
        }
        unsigned int w[4];
#pragma unroll
        for (int q = 0; q < 4; ++q)
            w[q] = (unsigned int)bytes[4*q] | ((unsigned int)bytes[4*q+1] << 8) |
                   ((unsigned int)bytes[4*q+2] << 16) | ((unsigned int)bytes[4*q+3] << 24);
        gtb[row]  = make_uint4(w[0], w[1], w[2], w[3]);
        has[row]  = (unsigned char)any;
    }
}

__device__ __forceinline__ v8i load_btile(const uint4* __restrict__ gtb,
                                          int jn, int col, unsigned int bmask) {
    uint4 w = gtb[jn + col];
    v8i b = {};
    b[0] = (int)(w.x & bmask); b[1] = (int)(w.y & bmask);
    b[2] = (int)(w.z & bmask); b[3] = (int)(w.w & bmask);
    return b;
}

__device__ __forceinline__ void consume_tile(
    const v8i& d, int jb, int col, unsigned int hj,
    const bool (&hasi)[8], const unsigned int (&rowbase)[8],
    u64 (&bestp)[8], u64 (&bestn)[8])
{
    const int j = jb + col;
    const bool hjj = hj != 0u;
#pragma unroll
    for (int r = 0; r < 8; ++r) {
        const bool share = d[r] > 0;
        const bool hi    = hasi[r];
        const bool pos   = hi ? share  : !hjj;
        const bool neg   = hi ? !share :  hjj;
        const unsigned int h  = mix32(rowbase[r] + (unsigned int)j);
        const unsigned int kp = h;
        const unsigned int kn = h * 0x9E3779B9u + 0x7F4A7C15u;
        const u64 cp = pos ? (((u64)kp << 32) | (unsigned int)j) : 0ull;
        const u64 cn = neg ? (((u64)kn << 32) | (unsigned int)j) : 0ull;
        bestp[r] = (cp > bestp[r]) ? cp : bestp[r];
        bestn[r] = (cn > bestn[r]) ? cn : bestn[r];
    }
}

__global__ __launch_bounds__(128) void trl_phase2_sample(
    const uint4* __restrict__ gtb, const unsigned char* __restrict__ has,
    int* __restrict__ pos_idx, int* __restrict__ neg_idx)
{
    const int lane  = threadIdx.x & 31;
    const int wave  = threadIdx.x >> 5;
    const int ibase = (blockIdx.x * 4 + wave) * 16;
    const int col   = lane & 15;
    const int half  = lane >> 4;
    const unsigned int bmask = half ? 0u : 0xFFFFFFFFu;

    uint4 aw = gtb[ibase + col];
    v8i a = {};
    a[0] = (int)(half ? 0u : aw.x);
    a[1] = (int)(half ? 0u : aw.y);
    a[2] = (int)(half ? 0u : aw.z);
    a[3] = (int)(half ? 0u : aw.w);

    int rowi[8]; unsigned int rowbase[8]; bool hasi[8];
#pragma unroll
    for (int r = 0; r < 8; ++r) {
        rowi[r]    = ibase + half * 8 + r;
        rowbase[r] = (unsigned int)rowi[r] * 8192u;
        hasi[r]    = has[rowi[r]] != 0;
    }

    u64 bestp[8], bestn[8];
#pragma unroll
    for (int r = 0; r < 8; ++r) { bestp[r] = (u64)(unsigned int)rowi[r]; bestn[r] = bestp[r]; }

    v8i b0 = load_btile(gtb, 0,  col, bmask);
    v8i b1 = load_btile(gtb, 16, col, bmask);
    unsigned int hj0 = has[col];
    unsigned int hj1 = has[16 + col];
    v8i cz0 = {}, cz1 = {};
    v8i d0 = __builtin_amdgcn_wmma_i32_16x16x64_iu8(false, a, false, b0, cz0, false, false);
    v8i d1 = __builtin_amdgcn_wmma_i32_16x16x64_iu8(false, a, false, b1, cz1, false, false);

    for (int jb = 0; jb < B_ROWS; jb += 32) {
        const int jn0 = (jb + 32) & (B_ROWS - 1);
        v8i bn0 = load_btile(gtb, jn0, col, bmask);
        const unsigned int hn0 = has[jn0 + col];

        consume_tile(d0, jb, col, hj0, hasi, rowbase, bestp, bestn);

        v8i czA = {};
        d0  = __builtin_amdgcn_wmma_i32_16x16x64_iu8(false, a, false, bn0, czA, false, false);
        hj0 = hn0;

        const int jn1 = (jb + 48) & (B_ROWS - 1);
        v8i bn1 = load_btile(gtb, jn1, col, bmask);
        const unsigned int hn1 = has[jn1 + col];

        consume_tile(d1, jb + 16, col, hj1, hasi, rowbase, bestp, bestn);

        v8i czB = {};
        d1  = __builtin_amdgcn_wmma_i32_16x16x64_iu8(false, a, false, bn1, czB, false, false);
        hj1 = hn1;
    }

#pragma unroll
    for (int r = 0; r < 8; ++r) {
#pragma unroll
        for (int off = 8; off; off >>= 1) {
            const u64 op = shfl_xor_u64(bestp[r], off, 16);
            const u64 on = shfl_xor_u64(bestn[r], off, 16);
            bestp[r] = (op > bestp[r]) ? op : bestp[r];
            bestn[r] = (on > bestn[r]) ? on : bestn[r];
        }
        if (col == 0) {
            pos_idx[rowi[r]] = (int)(unsigned int)bestp[r];
            neg_idx[rowi[r]] = (int)(unsigned int)bestn[r];
        }
    }
}

__global__ __launch_bounds__(128) void trl_phase3_loss(
    const float* __restrict__ pred, const float* __restrict__ norms,
    const int* __restrict__ pos_idx, const int* __restrict__ neg_idx,
    float* __restrict__ out)
{
    const int lane = threadIdx.x & 31;
    const int row  = blockIdx.x * 4 + (threadIdx.x >> 5);

    const int p = pos_idx[row];
    const int n = neg_idx[row];
    const float4* ar = (const float4*)(pred + (size_t)row * D_DIM);
    const float4* pr = (const float4*)(pred + (size_t)p * D_DIM);
    const float4* nr = (const float4*)(pred + (size_t)n * D_DIM);

    float dp = 0.f, dn = 0.f;
#pragma unroll
    for (int k = 0; k < 8; ++k) {
        float4 av = ar[lane + 32 * k];
        float4 pv = pr[lane + 32 * k];
        float4 nv = nr[lane + 32 * k];
        dp += av.x * pv.x + av.y * pv.y + av.z * pv.z + av.w * pv.w;
        dn += av.x * nv.x + av.y * nv.y + av.z * nv.z + av.w * nv.w;
    }
#pragma unroll
    for (int off = 16; off; off >>= 1) {
        dp += __shfl_xor(dp, off, 32);
        dn += __shfl_xor(dn, off, 32);
    }

    if (lane == 0) {
        const float na = norms[row];
        const float cp = dp / fmaxf(na * norms[p], EPS);
        const float cn = dn / fmaxf(na * norms[n], EPS);
        float v = cp - cn + MARGIN;
        v = fmaxf(v, 0.f);
        atomicAdd(out, v * (1.0f / (float)B_ROWS));
    }
}

extern "C" void kernel_launch(void* const* d_in, const int* in_sizes, int n_in,
                              void* d_out, int out_size, void* d_ws, size_t ws_size,
                              hipStream_t stream) {
    const float* pred = (const float*)d_in[0];
    const float* gt   = (const float*)d_in[1];

    char* ws = (char*)d_ws;
    uint4*         gtb     = (uint4*)ws;
    unsigned char* has     = (unsigned char*)(ws + 131072);
    float*         norms   = (float*)(ws + 139264);
    int*           pos_idx = (int*)(ws + 172032);
    int*           neg_idx = (int*)(ws + 204800);
    float*         out     = (float*)d_out;

    hipMemsetAsync(out, 0, sizeof(float), stream);

    trl_phase1_prep  <<<B_ROWS / 4,  128, 0, stream>>>(pred, gt, gtb, has, norms);
    trl_phase2_sample<<<B_ROWS / 64, 128, 0, stream>>>(gtb, has, pos_idx, neg_idx);
    trl_phase3_loss  <<<B_ROWS / 4,  128, 0, stream>>>(pred, norms, pos_idx, neg_idx, out);
}
